// GNNEncoder_32933809226061
// MI455X (gfx1250) — compile-verified, matrix operand order checked by reading, not run on hardware
//
#include <hip/hip_runtime.h>
#include <hip/hip_bf16.h>

// ---------------------------------------------------------------------------
// GIN encoder: 3x [agg(scatter-add) -> MLP(2x128 GEMM, ReLU)] -> graph pool
// MI455X/gfx1250: bf16 WMMA 16x16x32 for all 128x128 GEMMs.
// Round 2: branch-free WMMA GEMM; (h + agg) pre-add hoisted into a packed
// elementwise kernel that updates the bf16 feature buffer in place.
// ---------------------------------------------------------------------------

#define N_NODES   100000
#define N_EDGES   3200000
#define N_GRAPHS  1024
#define IN_DIM    13
#define HIDDEN    128

typedef __attribute__((ext_vector_type(16))) __bf16 v16bf;
typedef __attribute__((ext_vector_type(8)))  float  v8f;

union AFrag { v16bf v; unsigned short u[16]; uint4 q; };

__device__ __forceinline__ unsigned short f2bf(float f) {
    unsigned u = __builtin_bit_cast(unsigned, f);
    unsigned r = 0x7FFFu + ((u >> 16) & 1u);          // round-to-nearest-even
    return (unsigned short)((u + r) >> 16);
}
__device__ __forceinline__ float bf2f(unsigned short h) {
    return __builtin_bit_cast(float, (unsigned)h << 16);
}

// --------------------------- fp32 -> bf16 weight convert --------------------
__global__ void cvt_bf16_kernel(const float* __restrict__ in,
                                unsigned short* __restrict__ out, int n) {
    int i = blockIdx.x * blockDim.x + threadIdx.x;
    if (i < n) out[i] = f2bf(in[i]);
}

// --------------------------- layer-0 scatter (13 feats, fp32) ---------------
__global__ void scatter13_kernel(const float* __restrict__ x,
                                 const int* __restrict__ src,
                                 const int* __restrict__ dst,
                                 float* __restrict__ agg, int nE) {
    int e = blockIdx.x * blockDim.x + threadIdx.x;
    if (e >= nE) return;
    int s = src[e], d = dst[e];
    const float* xr = x + s * IN_DIM;
    float* ar = agg + d * IN_DIM;
#pragma unroll
    for (int k = 0; k < IN_DIM; ++k) atomicAdd(ar + k, xr[k]);
}

// --------------------------- layer-0 first MLP (K = 13) ---------------------
// h = relu((x + agg) @ w0a + b0a)  -> bf16
__global__ void mlp0a_kernel(const float* __restrict__ x,
                             const float* __restrict__ agg,
                             const float* __restrict__ w,   // [13][128]
                             const float* __restrict__ b,   // [128]
                             unsigned short* __restrict__ out) {
    __shared__ float wl[IN_DIM * HIDDEN];
    __shared__ float bl[HIDDEN];
    int tid = threadIdx.x;
    for (int i = tid; i < IN_DIM * HIDDEN; i += 256) wl[i] = w[i];
    if (tid < HIDDEN) bl[tid] = b[tid];
    __syncthreads();
    int idx = blockIdx.x * 256 + tid;                  // node*128 + col
    if (idx >= N_NODES * HIDDEN) return;
    int node = idx >> 7, col = idx & 127;
    const float* xr = x + node * IN_DIM;
    const float* ar = agg + node * IN_DIM;
    float acc = bl[col];
#pragma unroll
    for (int k = 0; k < IN_DIM; ++k) acc += (xr[k] + ar[k]) * wl[k * HIDDEN + col];
    out[idx] = f2bf(acc > 0.f ? acc : 0.f);
}

// --------------------------- in-place pre-add: h = bf16(f32(h) + agg) -------
// One thread handles 8 contiguous features (16 B bf16 load/store, 32 B f32).
__global__ void preadd_kernel(unsigned short* __restrict__ h,
                              const float* __restrict__ agg) {
    int g = blockIdx.x * blockDim.x + threadIdx.x;     // group of 8 elements
    if (g >= (N_NODES * HIDDEN) / 8) return;
    int base = g * 8;
    uint4 hv = *(const uint4*)(h + base);
    float4 a0 = *(const float4*)(agg + base);
    float4 a1 = *(const float4*)(agg + base + 4);
    unsigned w[4] = {hv.x, hv.y, hv.z, hv.w};
    float s[8];
    s[0] = bf2f((unsigned short)(w[0] & 0xffff)) + a0.x;
    s[1] = bf2f((unsigned short)(w[0] >> 16))    + a0.y;
    s[2] = bf2f((unsigned short)(w[1] & 0xffff)) + a0.z;
    s[3] = bf2f((unsigned short)(w[1] >> 16))    + a0.w;
    s[4] = bf2f((unsigned short)(w[2] & 0xffff)) + a1.x;
    s[5] = bf2f((unsigned short)(w[2] >> 16))    + a1.y;
    s[6] = bf2f((unsigned short)(w[3] & 0xffff)) + a1.z;
    s[7] = bf2f((unsigned short)(w[3] >> 16))    + a1.w;
    uint4 o;
    o.x = (unsigned)f2bf(s[0]) | ((unsigned)f2bf(s[1]) << 16);
    o.y = (unsigned)f2bf(s[2]) | ((unsigned)f2bf(s[3]) << 16);
    o.z = (unsigned)f2bf(s[4]) | ((unsigned)f2bf(s[5]) << 16);
    o.w = (unsigned)f2bf(s[6]) | ((unsigned)f2bf(s[7]) << 16);
    *(uint4*)(h + base) = o;
}

// --------------------------- bf16 WMMA GEMM: [M,128]@[128,128]+bias, ReLU ---
// Branch-free. One wave -> one 16x16 tile; 8 waves/block = 16x128 row slab.
__global__ void gemm128_wmma_kernel(const unsigned short* __restrict__ A,
                                    const unsigned short* __restrict__ W,
                                    const float* __restrict__ bias,
                                    unsigned short* __restrict__ Out) {
    __shared__ unsigned short Wt[HIDDEN * HIDDEN];     // transposed: Wt[n][k], 32 KB
    int tid = threadIdx.x;

    // stage W transposed into LDS (vector global loads, strided LDS stores)
    for (int i = tid * 8; i < HIDDEN * HIDDEN; i += 256 * 8) {
        uint4 v = *(const uint4*)(W + i);
        int k = i >> 7, n = i & 127;
        unsigned short e[8];
        e[0] = (unsigned short)(v.x & 0xffff); e[1] = (unsigned short)(v.x >> 16);
        e[2] = (unsigned short)(v.y & 0xffff); e[3] = (unsigned short)(v.y >> 16);
        e[4] = (unsigned short)(v.z & 0xffff); e[5] = (unsigned short)(v.z >> 16);
        e[6] = (unsigned short)(v.w & 0xffff); e[7] = (unsigned short)(v.w >> 16);
#pragma unroll
        for (int j = 0; j < 8; ++j) Wt[(n + j) * HIDDEN + k] = e[j];
    }
    __syncthreads();

    const int wave  = tid >> 5;            // N-tile 0..7
    const int lane  = tid & 31;
    const int l16   = lane & 15;
    const int kbase = (lane >> 4) * 8;     // 0 or 8
    const int n0    = wave * 16;
    const int rowM  = blockIdx.x * 16 + l16;

    const unsigned short* Arow = A + rowM * HIDDEN;
    const unsigned short* Bcol = Wt + (n0 + l16) * HIDDEN;

    v8f acc = {};
#pragma unroll
    for (int k0 = 0; k0 < HIDDEN; k0 += 32) {
        AFrag a, bfr;
        a.q   = *(const uint4*)(Arow + k0 + kbase);                 // K  .. K+7
        *(uint4*)(a.u + 8)   = *(const uint4*)(Arow + k0 + kbase + 16);
        bfr.q = *(const uint4*)(Bcol + k0 + kbase);
        *(uint4*)(bfr.u + 8) = *(const uint4*)(Bcol + k0 + kbase + 16);

        acc = __builtin_amdgcn_wmma_f32_16x16x32_bf16(
            /*neg_a=*/false, a.v, /*neg_b=*/false, bfr.v,
            /*c_mod=*/(short)0, acc, /*reuse_a=*/false, /*reuse_b=*/false);
    }

    const int n = n0 + l16;
    const float bv = bias[n];
    const int mBase = blockIdx.x * 16 + (lane >> 4) * 8;
#pragma unroll
    for (int r = 0; r < 8; ++r) {
        float v = acc[r] + bv;
        v = v > 0.f ? v : 0.f;                         // both GEMMs are ReLU'd
        Out[(mBase + r) * HIDDEN + n] = f2bf(v);
    }
}

// --------------------------- 128-feat edge scatter (bf16 -> fp32 atomics) ---
__global__ void scatter128_kernel(const unsigned short* __restrict__ h,
                                  const int* __restrict__ src,
                                  const int* __restrict__ dst,
                                  float* __restrict__ agg, int nE) {
    int gid = blockIdx.x * blockDim.x + threadIdx.x;   // edge*16 + group
    int e = gid >> 4, g = gid & 15;
    if (e >= nE) return;
    int s = src[e], d = dst[e];
    uint4 v = *(const uint4*)(h + s * HIDDEN + g * 8);
    float* q = agg + d * HIDDEN + g * 8;
    unsigned w[4] = {v.x, v.y, v.z, v.w};
#pragma unroll
    for (int j = 0; j < 4; ++j) {
        atomicAdd(q + 2 * j,     bf2f((unsigned short)(w[j] & 0xffff)));
        atomicAdd(q + 2 * j + 1, bf2f((unsigned short)(w[j] >> 16)));
    }
}

// --------------------------- global add pool --------------------------------
__global__ void pool_kernel(const unsigned short* __restrict__ h,
                            const int* __restrict__ batch,
                            float* __restrict__ out) {
    int idx = blockIdx.x * blockDim.x + threadIdx.x;   // node*128 + feat
    if (idx >= N_NODES * HIDDEN) return;
    int node = idx >> 7, f = idx & 127;
    int g = batch[node];
    atomicAdd(out + g * HIDDEN + f, bf2f(h[idx]));
}

// ---------------------------------------------------------------------------
extern "C" void kernel_launch(void* const* d_in, const int* in_sizes, int n_in,
                              void* d_out, int out_size, void* d_ws, size_t ws_size,
                              hipStream_t stream) {
    const float* x    = (const float*)d_in[0];
    const int*   edge = (const int*)d_in[1];
    const int*   bat  = (const int*)d_in[2];
    const float* w0a = (const float*)d_in[3],  *b0a = (const float*)d_in[4];
    const float* w0b = (const float*)d_in[5],  *b0b = (const float*)d_in[6];
    const float* w1a = (const float*)d_in[7],  *b1a = (const float*)d_in[8];
    const float* w1b = (const float*)d_in[9],  *b1b = (const float*)d_in[10];
    const float* w2a = (const float*)d_in[11], *b2a = (const float*)d_in[12];
    const float* w2b = (const float*)d_in[13], *b2b = (const float*)d_in[14];

    const int nE = in_sizes[1] / 2;
    const int* src = edge;
    const int* dst = edge + nE;

    char* ws = (char*)d_ws;
    float*          agg = (float*)ws;                                   // 51.2 MB
    unsigned short* hA  = (unsigned short*)(ws + 51200000);             // 25.6 MB
    unsigned short* hB  = (unsigned short*)(ws + 76800000);             // 25.6 MB
    unsigned short* wbf = (unsigned short*)(ws + 102400000);            // 160 KB
    unsigned short* w0b_bf = wbf;
    unsigned short* w1a_bf = wbf + 1 * HIDDEN * HIDDEN;
    unsigned short* w1b_bf = wbf + 2 * HIDDEN * HIDDEN;
    unsigned short* w2a_bf = wbf + 3 * HIDDEN * HIDDEN;
    unsigned short* w2b_bf = wbf + 4 * HIDDEN * HIDDEN;

    const int WN = HIDDEN * HIDDEN;
    cvt_bf16_kernel<<<(WN + 255) / 256, 256, 0, stream>>>(w0b, w0b_bf, WN);
    cvt_bf16_kernel<<<(WN + 255) / 256, 256, 0, stream>>>(w1a, w1a_bf, WN);
    cvt_bf16_kernel<<<(WN + 255) / 256, 256, 0, stream>>>(w1b, w1b_bf, WN);
    cvt_bf16_kernel<<<(WN + 255) / 256, 256, 0, stream>>>(w2a, w2a_bf, WN);
    cvt_bf16_kernel<<<(WN + 255) / 256, 256, 0, stream>>>(w2b, w2b_bf, WN);

    const int gemmBlocks = N_NODES / 16;                                // 6250
    const int nodeFeat   = N_NODES * HIDDEN;
    const int preBlocks  = (nodeFeat / 8 + 255) / 256;

    // ---- layer 0 ----
    hipMemsetAsync(agg, 0, (size_t)N_NODES * IN_DIM * sizeof(float), stream);
    scatter13_kernel<<<(nE + 255) / 256, 256, 0, stream>>>(x, src, dst, agg, nE);
    mlp0a_kernel<<<(nodeFeat + 255) / 256, 256, 0, stream>>>(x, agg, w0a, b0a, hA);
    gemm128_wmma_kernel<<<gemmBlocks, 256, 0, stream>>>(hA, w0b_bf, b0b, hB);

    // ---- layer 1 ----
    hipMemsetAsync(agg, 0, (size_t)N_NODES * HIDDEN * sizeof(float), stream);
    scatter128_kernel<<<(nE * 16 + 255) / 256, 256, 0, stream>>>(hB, src, dst, agg, nE);
    preadd_kernel<<<preBlocks, 256, 0, stream>>>(hB, agg);              // hB = pre1
    gemm128_wmma_kernel<<<gemmBlocks, 256, 0, stream>>>(hB, w1a_bf, b1a, hA);
    gemm128_wmma_kernel<<<gemmBlocks, 256, 0, stream>>>(hA, w1b_bf, b1b, hB);

    // ---- layer 2 ----
    hipMemsetAsync(agg, 0, (size_t)N_NODES * HIDDEN * sizeof(float), stream);
    scatter128_kernel<<<(nE * 16 + 255) / 256, 256, 0, stream>>>(hB, src, dst, agg, nE);
    preadd_kernel<<<preBlocks, 256, 0, stream>>>(hB, agg);              // hB = pre2
    gemm128_wmma_kernel<<<gemmBlocks, 256, 0, stream>>>(hB, w2a_bf, b2a, hA);
    gemm128_wmma_kernel<<<gemmBlocks, 256, 0, stream>>>(hA, w2b_bf, b2b, hB);

    // ---- pool ----
    hipMemsetAsync(d_out, 0, (size_t)N_GRAPHS * HIDDEN * sizeof(float), stream);
    pool_kernel<<<(nodeFeat + 255) / 256, 256, 0, stream>>>(hB, bat, (float*)d_out);
}
